// LSTMAttnDecoder_4337916969749
// MI455X (gfx1250) — matrix operand order repaired and checked by reading, not compiled, not run on hardware
//
#include <hip/hip_runtime.h>
#include <hip/hip_bf16.h>

// ---------------------------------------------------------------------------
// LSTM + attention decoder for MI455X (gfx1250, wave32, WMMA).
//   B=32, S=196, T=64, H=1024, V=32000
// Strategy:
//   - sequential recurrence (64 steps) with fp32 WMMA 16x16x4 GEMMs
//   - one large bf16 WMMA 16x16x32 GEMM for logits [B*T, V]
//   - LDS panel staging via GLOBAL_LOAD_ASYNC_TO_LDS_B128 (ASYNCcnt) when
//     the toolchain exposes the builtin; VGPR-copy fallback otherwise.
// ---------------------------------------------------------------------------

#define Bdim 32
#define Sdim 196
#define Tdim 64
#define Hdim 1024
#define Vdim 32000
#define H4   4096
#define MT   (Bdim * Tdim)   // 2048 rows of the logits GEMM

typedef __attribute__((ext_vector_type(2)))  float  v2f;
typedef __attribute__((ext_vector_type(8)))  float  v8f;
typedef __attribute__((ext_vector_type(16))) __bf16 v16bf;
typedef __attribute__((ext_vector_type(4)))  int    v4i;

#if defined(__has_builtin)
#  if __has_builtin(__builtin_amdgcn_global_load_async_to_lds_b128)
#    define HAVE_ASYNC_LDS 1
#  endif
#  if __has_builtin(__builtin_amdgcn_s_wait_asynccnt)
#    define HAVE_WAIT_ASYNC_BUILTIN 1
#  endif
#endif

// Pointer flavors for the async DMA builtin: global (AS1) src, LDS (AS3) dst.
typedef __attribute__((address_space(1))) v4i* gptr_v4i;
typedef __attribute__((address_space(3))) v4i* lptr_v4i;

__device__ __forceinline__ void async_copy_b128(const void* gsrc, void* ldst) {
#if defined(HAVE_ASYNC_LDS)
    __builtin_amdgcn_global_load_async_to_lds_b128(
        (gptr_v4i)gsrc, (lptr_v4i)ldst, 0, 0);
#else
    *(uint4*)ldst = *(const uint4*)gsrc;
#endif
}

__device__ __forceinline__ void wait_async_lds() {
#if defined(HAVE_ASYNC_LDS)
#  if defined(HAVE_WAIT_ASYNC_BUILTIN)
    __builtin_amdgcn_s_wait_asynccnt(0);
#  else
    asm volatile("s_wait_asynccnt 0x0" ::: "memory");
#  endif
#endif
}

__device__ __forceinline__ unsigned short f2bf(float x) {
    unsigned int u = __float_as_uint(x);
    unsigned int r = (u + 0x7FFFu + ((u >> 16) & 1u)) >> 16;  // round-nearest-even
    return (unsigned short)r;
}

__device__ __forceinline__ float sigf(float x) {
    return 1.0f / (1.0f + __expf(-x));
}

// ---------------------------------------------------------------------------
// fp32 -> bf16 bulk convert (W_out, done once per launch; 131MB -> 64MB)
// ---------------------------------------------------------------------------
__global__ void f32_to_bf16_kernel(const float* __restrict__ in,
                                   unsigned short* __restrict__ out, long n) {
    long i = (long)blockIdx.x * blockDim.x + threadIdx.x;
    long stride = (long)gridDim.x * blockDim.x;
    for (; i < n; i += stride) out[i] = f2bf(in[i]);
}

// ---------------------------------------------------------------------------
// Embedding gather: emb_all[b*T + t, :] = emb_table[captions[b, t], :]
// ---------------------------------------------------------------------------
__global__ void gather_emb_kernel(const int* __restrict__ cap,
                                  const float* __restrict__ emb_table,
                                  float* __restrict__ emb_all) {
    int bt = blockIdx.x;                       // b*T + t
    long v = (long)cap[bt];
    const float* src = emb_table + v * Hdim;
    float* dst = emb_all + (long)bt * Hdim;
    for (int i = threadIdx.x; i < Hdim; i += blockDim.x) dst[i] = src[i];
}

__global__ void zero_f32_kernel(float* p, int n) {
    int i = blockIdx.x * blockDim.x + threadIdx.x;
    if (i < n) p[i] = 0.0f;
}

// ---------------------------------------------------------------------------
// Generic small GEMM with fp32 WMMA:  C[32, N] (+)= A[32, K] @ W[N, K]^T
// One 16x16 tile per wave, blocks of 4 waves. M fixed at 32 (grid.y = 2).
// A row stride = lda, W row stride = ldw (lets us slice W_ih columns).
// fp32 A 16x4 layout: lane&15 = M row; lane>>4 selects K pair {0,1} vs {2,3}.
// fp32 B 4x16 layout: lane&15 = N col; VGPR r holds K = r + 2*(lane>>4).
// ---------------------------------------------------------------------------
__global__ __launch_bounds__(128)
void gemm32_f32_wmma(const float* __restrict__ A, int lda,
                     const float* __restrict__ W, int ldw,
                     float* __restrict__ C, int ldc,
                     int N, int K, int acc_flag) {
    int lane = threadIdx.x & 31;
    int wid  = threadIdx.x >> 5;
    int n_base = blockIdx.x * 64 + wid * 16;
    int m_base = blockIdx.y * 16;
    if (n_base >= N) return;

    int r15 = lane & 15;
    int sel = lane >> 4;

    const float* arow = A + (long)(m_base + r15) * lda;
    const float* wrow = W + (long)(n_base + r15) * ldw;

    v8f acc = {};
    if (acc_flag) {
#pragma unroll
        for (int r = 0; r < 8; ++r)
            acc[r] = C[(long)(m_base + r + sel * 8) * ldc + n_base + r15];
    }

    for (int k0 = 0; k0 < K; k0 += 4) {
        v2f a = *(const v2f*)(arow + k0 + sel * 2);
        v2f b = *(const v2f*)(wrow + k0 + sel * 2);
        acc = __builtin_amdgcn_wmma_f32_16x16x4_f32(false, a, false, b,
                                                    (short)0, acc, false, false);
    }

#pragma unroll
    for (int r = 0; r < 8; ++r)
        C[(long)(m_base + r + sel * 8) * ldc + n_base + r15] = acc[r];
}

// ---------------------------------------------------------------------------
// Fused attention: scores = q . memory[b,s,:], softmax over s, ctx = w . memory
// One block per batch row; 8 waves; wave-per-score with lane reduction.
// q staged to LDS with async b128 copies (one per thread).
// ---------------------------------------------------------------------------
__global__ __launch_bounds__(256)
void attn_kernel(const float* __restrict__ q,
                 const float* __restrict__ memory,
                 float* __restrict__ ctx) {
    __shared__ __align__(16) float qs[Hdim];
    __shared__ float sc[Sdim];
    __shared__ float red[2];

    int b = blockIdx.x;
    int tid = threadIdx.x;
    int lane = tid & 31;
    int wid  = tid >> 5;
    const float* mem = memory + (long)b * Sdim * Hdim;

    // Stage q (4KB) into LDS: 256 threads x one b128 async copy.
    async_copy_b128((const uint4*)(q + (long)b * Hdim) + tid,
                    (uint4*)qs + tid);
    wait_async_lds();
    __syncthreads();

    for (int s = wid; s < Sdim; s += 8) {
        const float* mrow = mem + (long)s * Hdim;
        float p = 0.0f;
        for (int h = lane; h < Hdim; h += 32) p += qs[h] * mrow[h];
#pragma unroll
        for (int off = 16; off > 0; off >>= 1) p += __shfl_xor(p, off, 32);
        if (lane == 0) sc[s] = p;
    }
    __syncthreads();

    if (tid == 0) {
        float m = sc[0];
        for (int s = 1; s < Sdim; ++s) m = fmaxf(m, sc[s]);
        red[0] = m;
    }
    __syncthreads();
    if (tid < Sdim) sc[tid] = __expf(sc[tid] - red[0]);
    __syncthreads();
    if (tid == 0) {
        float ssum = 0.0f;
        for (int s = 0; s < Sdim; ++s) ssum += sc[s];
        red[1] = 1.0f / ssum;
    }
    __syncthreads();

    float inv = red[1];
    for (int h = tid; h < Hdim; h += 256) {
        float a = 0.0f;
        for (int s = 0; s < Sdim; ++s) a += sc[s] * mem[(long)s * Hdim + h];
        ctx[(long)b * Hdim + h] = a * inv;
    }
}

// ---------------------------------------------------------------------------
// LSTM pointwise: gates -> h_new, c_new; also record bf16 h into h_all row b*T+t
// ---------------------------------------------------------------------------
__global__ __launch_bounds__(256)
void lstm_kernel(const float* __restrict__ gates,
                 const float* __restrict__ b_ih, const float* __restrict__ b_hh,
                 float* __restrict__ h_cur, float* __restrict__ c_cur,
                 unsigned short* __restrict__ h_all, int t) {
    int b = blockIdx.x;
    const float* g = gates + (long)b * H4;
    for (int n = threadIdx.x; n < Hdim; n += 256) {
        float gi = g[n]            + b_ih[n]            + b_hh[n];
        float gf = g[Hdim + n]     + b_ih[Hdim + n]     + b_hh[Hdim + n];
        float gg = g[2 * Hdim + n] + b_ih[2 * Hdim + n] + b_hh[2 * Hdim + n];
        float go = g[3 * Hdim + n] + b_ih[3 * Hdim + n] + b_hh[3 * Hdim + n];
        float i_ = sigf(gi), f_ = sigf(gf), o_ = sigf(go);
        float g_ = tanhf(gg);
        long idx = (long)b * Hdim + n;
        float cn = f_ * c_cur[idx] + i_ * g_;
        float hn = o_ * tanhf(cn);
        c_cur[idx] = cn;
        h_cur[idx] = hn;
        h_all[((long)b * Tdim + t) * Hdim + n] = f2bf(hn);
    }
}

// ---------------------------------------------------------------------------
// Big logits GEMM (bf16 WMMA 16x16x32):
//   out[2048, 32000] = h_all_bf16[2048,1024] @ W_out_bf16[32000,1024]^T + b_out
// Block: 64 M-rows x 256 N-cols, full K panel of A staged in 128KB LDS via
// GLOBAL_LOAD_ASYNC_TO_LDS_B128 (ASYNCcnt-tracked DMA path on CDNA5).
// 8 waves: wave w -> m-tile (w&3), 8 consecutive n-tiles in half (w>>2).
// 16-bit A frag: lane&15 = row, chunks at k0+sel*8 and k0+16+sel*8.
// 16-bit B frag: lane&15 = col n, chunks at k0+sel*8 and k0+16+sel*8.
// ---------------------------------------------------------------------------
__global__ __launch_bounds__(256)
void logits_gemm_bf16(const unsigned short* __restrict__ Abf,
                      const unsigned short* __restrict__ Wbf,
                      const float* __restrict__ b_out,
                      float* __restrict__ out) {
    extern __shared__ unsigned short sA[];   // 64 * 1024 bf16 = 128 KB
    int tid = threadIdx.x;
    long m_panel = (long)blockIdx.y * 64;

    // Stage A panel (64 x 1024 bf16) into LDS: async memory->LDS b128 copies,
    // 32 per thread; single wait + barrier before math.
    {
        const uint4* src = (const uint4*)(Abf + m_panel * Hdim);
        uint4* dst = (uint4*)sA;
#pragma unroll 4
        for (int i = tid; i < 64 * Hdim / 8; i += 256)
            async_copy_b128(src + i, dst + i);
        wait_async_lds();
    }
    __syncthreads();

    int lane = tid & 31;
    int wid  = tid >> 5;
    int r15  = lane & 15;
    int sel  = lane >> 4;
    int mt   = wid & 3;                    // which 16-row tile in the panel
    int n_base = blockIdx.x * 256 + (wid >> 2) * 128;

    const unsigned short* a_row = sA + (mt * 16 + r15) * Hdim;

    v8f acc[8] = {};

    for (int k0 = 0; k0 < Hdim; k0 += 32) {
        union { v16bf v; uint4 u[2]; } af;
        af.u[0] = *(const uint4*)(a_row + k0 + sel * 8);
        af.u[1] = *(const uint4*)(a_row + k0 + 16 + sel * 8);
#pragma unroll
        for (int i = 0; i < 8; ++i) {
            const unsigned short* b_row =
                Wbf + (long)(n_base + i * 16 + r15) * Hdim;
            union { v16bf v; uint4 u[2]; } bfm;
            bfm.u[0] = *(const uint4*)(b_row + k0 + sel * 8);
            bfm.u[1] = *(const uint4*)(b_row + k0 + 16 + sel * 8);
            acc[i] = __builtin_amdgcn_wmma_f32_16x16x32_bf16(
                false, af.v, false, bfm.v, (short)0, acc[i], false, false);
        }
    }

#pragma unroll
    for (int i = 0; i < 8; ++i) {
        int n = n_base + i * 16 + r15;
        float bo = b_out[n];
#pragma unroll
        for (int r = 0; r < 8; ++r) {
            long m = m_panel + mt * 16 + r + sel * 8;
            out[m * (long)Vdim + n] = acc[i][r] + bo;
        }
    }
}

// ---------------------------------------------------------------------------
// Final states copy: out_tail[0:BH] = h, out_tail[BH:2BH] = c
// ---------------------------------------------------------------------------
__global__ void copy_states_kernel(const float* __restrict__ h_cur,
                                   const float* __restrict__ c_cur,
                                   float* __restrict__ out_tail) {
    int i = blockIdx.x * blockDim.x + threadIdx.x;
    int n = Bdim * Hdim;
    if (i < n) {
        out_tail[i] = h_cur[i];
        out_tail[n + i] = c_cur[i];
    }
}

// ---------------------------------------------------------------------------
// Host driver
// ---------------------------------------------------------------------------
extern "C" void kernel_launch(void* const* d_in, const int* in_sizes, int n_in,
                              void* d_out, int out_size, void* d_ws, size_t ws_size,
                              hipStream_t stream) {
    const float* memory    = (const float*)d_in[0];
    const int*   captions  = (const int*)d_in[1];
    const float* emb_table = (const float*)d_in[2];
    const float* attn_W    = (const float*)d_in[3];
    const float* W_ih      = (const float*)d_in[4];
    const float* W_hh      = (const float*)d_in[5];
    const float* b_ih      = (const float*)d_in[6];
    const float* b_hh      = (const float*)d_in[7];
    const float* W_out     = (const float*)d_in[8];
    const float* b_out     = (const float*)d_in[9];
    float* out = (float*)d_out;

    char* ws = (char*)d_ws;
    size_t off = 0;
    auto alloc = [&](size_t bytes) -> void* {
        void* p = ws + off;
        off = (off + bytes + 255) & ~(size_t)255;
        return p;
    };
    unsigned short* Wbf     = (unsigned short*)alloc((size_t)Vdim * Hdim * 2); // 64 MB
    unsigned short* h_all   = (unsigned short*)alloc((size_t)MT * Hdim * 2);   //  4 MB
    float*          emb_all = (float*)alloc((size_t)MT * Hdim * 4);            //  8 MB
    float*          h_cur   = (float*)alloc((size_t)Bdim * Hdim * 4);
    float*          c_cur   = (float*)alloc((size_t)Bdim * Hdim * 4);
    float*          q_buf   = (float*)alloc((size_t)Bdim * Hdim * 4);
    float*          ctx     = (float*)alloc((size_t)Bdim * Hdim * 4);
    float*          gates   = (float*)alloc((size_t)Bdim * H4 * 4);

    // One-time per launch: bf16 weights, embedding gather, zero states.
    f32_to_bf16_kernel<<<4096, 256, 0, stream>>>(W_out, Wbf, (long)Vdim * Hdim);
    gather_emb_kernel<<<MT, 256, 0, stream>>>(captions, emb_table, emb_all);
    zero_f32_kernel<<<(Bdim * Hdim + 255) / 256, 256, 0, stream>>>(h_cur, Bdim * Hdim);
    zero_f32_kernel<<<(Bdim * Hdim + 255) / 256, 256, 0, stream>>>(c_cur, Bdim * Hdim);

    // Sequential recurrence: stream ordering provides the per-step dependency.
    for (int t = 0; t < Tdim; ++t) {
        // q = h @ attn_W^T
        gemm32_f32_wmma<<<dim3(Hdim / 64, 2), 128, 0, stream>>>(
            h_cur, Hdim, attn_W, Hdim, q_buf, Hdim, Hdim, Hdim, 0);
        // attention softmax + context
        attn_kernel<<<Bdim, 256, 0, stream>>>(q_buf, memory, ctx);
        // gates = emb_t @ W_ih[:, :H]^T
        gemm32_f32_wmma<<<dim3(H4 / 64, 2), 128, 0, stream>>>(
            emb_all + (long)t * Hdim, Tdim * Hdim, W_ih, 2 * Hdim,
            gates, H4, H4, Hdim, 0);
        // gates += ctx @ W_ih[:, H:]^T
        gemm32_f32_wmma<<<dim3(H4 / 64, 2), 128, 0, stream>>>(
            ctx, Hdim, W_ih + Hdim, 2 * Hdim, gates, H4, H4, Hdim, 1);
        // gates += h @ W_hh^T
        gemm32_f32_wmma<<<dim3(H4 / 64, 2), 128, 0, stream>>>(
            h_cur, Hdim, W_hh, Hdim, gates, H4, H4, Hdim, 1);
        // pointwise LSTM cell; records bf16 h_t into h_all
        lstm_kernel<<<Bdim, 256, 0, stream>>>(gates, b_ih, b_hh,
                                              h_cur, c_cur, h_all, t);
    }

    // Big parallel vocab projection: [2048, 1024] @ [1024, 32000]
    logits_gemm_bf16<<<dim3(Vdim / 256, MT / 64), 256,
                       64 * Hdim * sizeof(unsigned short), stream>>>(
        h_all, Wbf, b_out, out);

    // Final (h, c) states after the logits block.
    copy_states_kernel<<<(Bdim * Hdim + 255) / 256, 256, 0, stream>>>(
        h_cur, c_cur, out + (long)MT * Vdim);
}
